// MultiHeadAttention_14053132992864
// MI455X (gfx1250) — compile-verified, matrix operand order checked by reading, not run on hardware
//
#include <hip/hip_runtime.h>
#include <hip/hip_bf16.h>
#include <cstdint>
#include <cstddef>

// Problem constants (fixed by the reference)
#define BATCH   4
#define T_SEQ   1024
#define D_MODEL 1024
#define NHEAD   16
#define DH      64
#define ATTN_SCALE 0.35355339059327373f   // 64^-0.25

typedef __attribute__((ext_vector_type(16))) __bf16 v16bf;
typedef __attribute__((ext_vector_type(8)))  float  v8f;
typedef __attribute__((ext_vector_type(4)))  unsigned int v4u;
typedef __attribute__((ext_vector_type(8)))  int    v8i32;
typedef __attribute__((ext_vector_type(4)))  int    v4i32;

#if __has_builtin(__builtin_amdgcn_tensor_load_to_lds)
#define USE_TDM 1
#else
#define USE_TDM 0
#endif

// ---------------------------------------------------------------------------
// helpers
// ---------------------------------------------------------------------------
__device__ __forceinline__ uint16_t f2bf(float f) {
  union { float f; uint32_t u; } c; c.f = f;
  uint32_t u = c.u;
  uint32_t r = u + 0x7fffu + ((u >> 16) & 1u);   // round-to-nearest-even
  return (uint16_t)(r >> 16);
}

struct Frag16 {
  union { v16bf v; uint4 q[2]; uint32_t u[8]; };
};

// NT fragment load for v_wmma_f32_16x16x32_bf16.
// A (16xK) and B (Kx16; B column j == memory row j for NT) share this layout:
//   lane L holds row (L&15); K chunks at (L>>4)*8 and 16+(L>>4)*8.
__device__ __forceinline__ v16bf load_nt_frag(const uint16_t* base, int ld) {
  const int lane = threadIdx.x & 31;
  const int row  = lane & 15;
  const int ks   = (lane >> 4) << 3;
  const uint16_t* p = base + (size_t)row * ld + ks;
  Frag16 f;
  f.q[0] = *(const uint4*)(p);        // K = ks .. ks+7        (b128)
  f.q[1] = *(const uint4*)(p + 16);   // K = 16+ks .. 16+ks+7  (b128)
  return f.v;
}

// Same fragment layout, but from an LDS-resident [16+][32] row-major tile.
__device__ __forceinline__ v16bf load_lds_frag(const uint16_t* base) {
  const int lane = threadIdx.x & 31;
  const uint16_t* p = base + (lane & 15) * 32 + ((lane >> 4) << 3);
  Frag16 f;
  f.q[0] = *(const uint4*)(p);
  f.q[1] = *(const uint4*)(p + 16);
  return f.v;
}

#if USE_TDM
// TDM: async-load a 64-row x 32-col bf16 tile (row stride = ld elements) from
// global memory into LDS at byte offset lds_off. 2D D# descriptor, count=1.
// This toolchain's builtin takes 6 args:
//   (uint32x4 g0, int32x8 g1, int32x4 g2, int32x4 g3, int32x8 g4, i32 cpol)
__device__ __forceinline__ void tdm_load_tile_64x32(uint32_t lds_off,
                                                    const uint16_t* gsrc,
                                                    uint32_t ld_elems) {
  const uint64_t ga = (uint64_t)(uintptr_t)gsrc;
  v4u g0;
  g0[0] = 1u;                                                  // count=1 (valid user D#)
  g0[1] = lds_off;                                             // lds_addr (bytes)
  g0[2] = (uint32_t)ga;                                        // global_addr[31:0]
  g0[3] = ((uint32_t)(ga >> 32) & 0x01FFFFFFu) | (2u << 30);   // addr[56:32] | type=2
  const uint32_t DIM = 1u << 20;    // generous tensor bounds (tile is in-range)
  v8i32 g1;
  g1[0] = (int)(1u << 16);                                      // data_size=1 -> 2 bytes
  g1[1] = (int)((DIM & 0xFFFFu) << 16);                         // tensor_dim0[15:0]
  g1[2] = (int)(((DIM >> 16) & 0xFFFFu) | ((DIM & 0xFFFFu) << 16)); // dim0 hi | dim1 lo
  g1[3] = (int)(((DIM >> 16) & 0xFFFFu) | (32u << 16));         // dim1 hi | tile_dim0=32
  g1[4] = (int)(64u);                                           // tile_dim1=64, tile_dim2=0
  g1[5] = (int)ld_elems;                                        // tensor_dim0_stride[31:0]
  g1[6] = 0;                                                    // stride0 hi | stride1 lo
  g1[7] = 0;                                                    // stride1 hi
  const v4i32 z4 = {0, 0, 0, 0};
  const v8i32 z8 = {0, 0, 0, 0, 0, 0, 0, 0};
  __builtin_amdgcn_tensor_load_to_lds(g0, g1, z4, z4, z8, 0);
}
#endif

// ---------------------------------------------------------------------------
// fp32 -> bf16 pack
// ---------------------------------------------------------------------------
__global__ void __launch_bounds__(256)
pack_bf16(const float* __restrict__ in, uint16_t* __restrict__ out, int n) {
  int i = blockIdx.x * 256 + threadIdx.x;
  if (i < n) out[i] = f2bf(in[i]);
}

// ---------------------------------------------------------------------------
// Generic NT GEMM:  C[m,n] = alpha * ( sum_k A[m,k]*B[n,k] + bias[n] )
// A: [M,K] bf16 row-major, B: [N,K] bf16 row-major, C: [M,N] (bf16 or f32)
// One wave computes a 64x64 tile (4x4 grid of 16x16 WMMA tiles).
// B tiles are staged global->LDS by the Tensor Data Mover (double-buffered);
// A fragments load directly from global (hits 192MB L2).
// ---------------------------------------------------------------------------
template<bool OUT_F32, bool HAS_BIAS>
__global__ void __launch_bounds__(256)
gemm_nt(const uint16_t* __restrict__ A, const uint16_t* __restrict__ B,
        const float* __restrict__ bias, float alpha,
        void* __restrict__ Cout, int M, int N, int K) {
  const int lane = threadIdx.x & 31;
  const int w    = threadIdx.x >> 5;
  const int wid  = __builtin_amdgcn_readfirstlane(blockIdx.x * 8 + w);
  const int ntiles = N >> 6;
  const int tm = (wid / ntiles) << 6;
  const int tn = (wid % ntiles) << 6;
  if (tm >= M) return;

  v8f c[4][4] = {};

#if USE_TDM
  __shared__ uint16_t bs[8][2][64 * 32];     // 8 waves x double buffer x 4KB
  const uint32_t lds0 = __builtin_amdgcn_readfirstlane((uint32_t)(uintptr_t)&bs[w][0][0]);
  const uint32_t lds1 = __builtin_amdgcn_readfirstlane((uint32_t)(uintptr_t)&bs[w][1][0]);
  const uint16_t* Bt = B + (size_t)tn * K;
  tdm_load_tile_64x32(lds0, Bt, (uint32_t)K);          // prologue: k = 0 tile
#endif

  for (int k = 0; k < K; k += 32) {
    v16bf a[4], b[4];
#pragma unroll
    for (int i = 0; i < 4; ++i) a[i] = load_nt_frag(A + (size_t)(tm + 16 * i) * K + k, K);

#if USE_TDM
    const int cur = (k >> 5) & 1;
    if (k + 32 < K) {
      tdm_load_tile_64x32(cur ? lds0 : lds1, Bt + k + 32, (uint32_t)K);
      __builtin_amdgcn_s_wait_tensorcnt((short)1);     // current buffer complete
    } else {
      __builtin_amdgcn_s_wait_tensorcnt((short)0);
    }
    __asm__ volatile("" ::: "memory");                 // LDS now holds the tile
    const uint16_t* bsrc = &bs[w][cur][0];
#pragma unroll
    for (int j = 0; j < 4; ++j) b[j] = load_lds_frag(bsrc + (16 * j) * 32);
#else
#pragma unroll
    for (int j = 0; j < 4; ++j) b[j] = load_nt_frag(B + (size_t)(tn + 16 * j) * K + k, K);
#endif

#pragma unroll
    for (int i = 0; i < 4; ++i)
#pragma unroll
      for (int j = 0; j < 4; ++j)
        c[i][j] = __builtin_amdgcn_wmma_f32_16x16x32_bf16(
            false, a[i], false, b[j], (short)0, c[i][j], false, false);
  }

  const int mo = (lane >> 4) << 3;   // M offset within 16x16: VGPR r -> M = r + mo
  const int nn = lane & 15;          // N within 16x16
#pragma unroll
  for (int j = 0; j < 4; ++j) {
    const int n = tn + 16 * j + nn;
    const float bv = HAS_BIAS ? bias[n] : 0.0f;
#pragma unroll
    for (int i = 0; i < 4; ++i) {
#pragma unroll
      for (int r = 0; r < 8; ++r) {
        const int m = tm + 16 * i + mo + r;
        const float val = (c[i][j][r] + bv) * alpha;
        if (OUT_F32) ((float*)Cout)[(size_t)m * N + n] = val;
        else         ((uint16_t*)Cout)[(size_t)m * N + n] = f2bf(val);
      }
    }
  }
}

// ---------------------------------------------------------------------------
// Attention scores: qk[b,h,ti,tj] = q.kT + mask[ti,tj]   (fp32, into d_out)
// q,k: bf16 [B,T,D], head h = columns h*64..h*64+63, scale already folded in.
// grid: (32, B*H), one wave = 64x64 tile of the 1024x1024 score matrix.
// ---------------------------------------------------------------------------
__global__ void __launch_bounds__(256)
attn_scores(const uint16_t* __restrict__ q, const uint16_t* __restrict__ k,
            const float* __restrict__ mask, float* __restrict__ qk) {
  const int lane = threadIdx.x & 31;
  const int bh   = blockIdx.y;
  const int b    = bh >> 4;
  const int h    = bh & 15;
  const int wid  = blockIdx.x * 8 + (threadIdx.x >> 5);  // 0..255
  const int tm   = (wid >> 4) << 6;
  const int tn   = (wid & 15) << 6;

  const uint16_t* qb = q + (size_t)b * T_SEQ * D_MODEL + h * DH;
  const uint16_t* kb = k + (size_t)b * T_SEQ * D_MODEL + h * DH;

  v8f c[4][4] = {};
#pragma unroll
  for (int kk = 0; kk < DH; kk += 32) {
    v16bf a[4], bb[4];
#pragma unroll
    for (int i = 0; i < 4; ++i) a[i]  = load_nt_frag(qb + (size_t)(tm + 16 * i) * D_MODEL + kk, D_MODEL);
#pragma unroll
    for (int j = 0; j < 4; ++j) bb[j] = load_nt_frag(kb + (size_t)(tn + 16 * j) * D_MODEL + kk, D_MODEL);
#pragma unroll
    for (int i = 0; i < 4; ++i)
#pragma unroll
      for (int j = 0; j < 4; ++j)
        c[i][j] = __builtin_amdgcn_wmma_f32_16x16x32_bf16(
            false, a[i], false, bb[j], (short)0, c[i][j], false, false);
  }

  float* out = qk + (size_t)bh * T_SEQ * T_SEQ;
  const int mo = (lane >> 4) << 3;
  const int nn = lane & 15;
#pragma unroll
  for (int i = 0; i < 4; ++i)
#pragma unroll
    for (int j = 0; j < 4; ++j) {
      const int n = tn + 16 * j + nn;
#pragma unroll
      for (int r = 0; r < 8; ++r) {
        const int m = tm + 16 * i + mo + r;
        out[(size_t)m * T_SEQ + n] = c[i][j][r] + mask[(size_t)m * T_SEQ + n];
      }
    }
}

// ---------------------------------------------------------------------------
// v [B,T,D] bf16 -> vt [B,H,DH,T] bf16  (so PV GEMM is NT-shaped)
// ---------------------------------------------------------------------------
__global__ void __launch_bounds__(256)
transpose_v(const uint16_t* __restrict__ v, uint16_t* __restrict__ vt) {
  const int idx = blockIdx.x * 256 + threadIdx.x;     // over B*T*D
  const int col = idx & (D_MODEL - 1);
  const int j   = (idx >> 10) & (T_SEQ - 1);
  const int b   = idx >> 20;
  const int h   = col >> 6;
  const int d   = col & (DH - 1);
  vt[((((size_t)b * NHEAD + h) * DH + d) << 10) + j] = v[idx];
}

__device__ __forceinline__ uint32_t pack2exp(float a, float b, float mx, float inv) {
  const uint32_t lo = f2bf(__expf(a - mx) * inv);
  const uint32_t hi = f2bf(__expf(b - mx) * inv);
  return lo | (hi << 16);
}

// ---------------------------------------------------------------------------
// Fused softmax + (P @ V):  one wave per 16-row tile of one (b,h).
// Pass 1: single-pass ONLINE max+sum over the fp32 qk row (each row is split
//         across lane L and L+16), saving one full 256MB read of qk.
// Pass 2: build bf16 prob A-fragments in-register, WMMA against vt fragments.
// ---------------------------------------------------------------------------
__global__ void __launch_bounds__(256)
softmax_wv(const float* __restrict__ qk, const uint16_t* __restrict__ vt,
           uint16_t* __restrict__ wv) {
  const int lane = threadIdx.x & 31;
  const int wid  = blockIdx.x * 8 + (threadIdx.x >> 5);  // 0..4095
  const int bh   = wid >> 6;
  const int t0   = (wid & 63) << 4;
  const int b    = bh >> 4;
  const int h    = bh & 15;
  const int row  = lane & 15;
  const int half = lane >> 4;

  const float* rowbase = qk + ((size_t)bh * T_SEQ + t0 + row) * T_SEQ;
  const float* rowp    = rowbase + half * 512;

  // online max + sum(exp) in a single pass over 512 elements per lane
  float mx = -3.0e38f;
  float s  = 0.0f;
  for (int i = 0; i < 512; i += 4) {
    const float4 x = *(const float4*)(rowp + i);
    const float cm = fmaxf(fmaxf(x.x, x.y), fmaxf(x.z, x.w));
    const float mn = fmaxf(mx, cm);
    s = s * __expf(mx - mn)
        + __expf(x.x - mn) + __expf(x.y - mn)
        + __expf(x.z - mn) + __expf(x.w - mn);
    mx = mn;
  }
  // combine the two half-rows (lane L <-> lane L+16)
  const float omx = __shfl_xor(mx, 16, 32);
  const float os  = __shfl_xor(s, 16, 32);
  const float gm  = fmaxf(mx, omx);
  s  = s * __expf(mx - gm) + os * __expf(omx - gm);
  mx = gm;
  const float inv = 1.0f / s;

  const int ks = half << 3;
  const uint16_t* vtb = vt + (size_t)bh * DH * T_SEQ;

  v8f acc[4] = {};
  for (int j0 = 0; j0 < T_SEQ; j0 += 32) {
    // A fragment: softmax probs for K = j0..j0+31, packed in WMMA lane layout
    const float* p0 = rowbase + j0 + ks;
    const float4 x0 = *(const float4*)(p0);
    const float4 x1 = *(const float4*)(p0 + 4);
    const float4 x2 = *(const float4*)(p0 + 16);
    const float4 x3 = *(const float4*)(p0 + 20);
    Frag16 af;
    af.u[0] = pack2exp(x0.x, x0.y, mx, inv);
    af.u[1] = pack2exp(x0.z, x0.w, mx, inv);
    af.u[2] = pack2exp(x1.x, x1.y, mx, inv);
    af.u[3] = pack2exp(x1.z, x1.w, mx, inv);
    af.u[4] = pack2exp(x2.x, x2.y, mx, inv);
    af.u[5] = pack2exp(x2.z, x2.w, mx, inv);
    af.u[6] = pack2exp(x3.x, x3.y, mx, inv);
    af.u[7] = pack2exp(x3.z, x3.w, mx, inv);

    v8f a0 = acc[0], a1 = acc[1], a2 = acc[2], a3 = acc[3];
    v16bf b0 = load_nt_frag(vtb + (size_t)(0)  * T_SEQ + j0, T_SEQ);
    v16bf b1 = load_nt_frag(vtb + (size_t)(16) * T_SEQ + j0, T_SEQ);
    v16bf b2 = load_nt_frag(vtb + (size_t)(32) * T_SEQ + j0, T_SEQ);
    v16bf b3 = load_nt_frag(vtb + (size_t)(48) * T_SEQ + j0, T_SEQ);
    acc[0] = __builtin_amdgcn_wmma_f32_16x16x32_bf16(false, af.v, false, b0, (short)0, a0, false, false);
    acc[1] = __builtin_amdgcn_wmma_f32_16x16x32_bf16(false, af.v, false, b1, (short)0, a1, false, false);
    acc[2] = __builtin_amdgcn_wmma_f32_16x16x32_bf16(false, af.v, false, b2, (short)0, a2, false, false);
    acc[3] = __builtin_amdgcn_wmma_f32_16x16x32_bf16(false, af.v, false, b3, (short)0, a3, false, false);
  }

  // store wv[b, t0+.., h*64+..] as bf16
#pragma unroll
  for (int j = 0; j < 4; ++j) {
    const int col = h * DH + 16 * j + row;
#pragma unroll
    for (int r = 0; r < 8; ++r) {
      const int m = t0 + (half << 3) + r;
      wv[((size_t)b * T_SEQ + m) * D_MODEL + col] = f2bf(acc[j][r]);
    }
  }
}

// ---------------------------------------------------------------------------
// launch
// ---------------------------------------------------------------------------
extern "C" void kernel_launch(void* const* d_in, const int* in_sizes, int n_in,
                              void* d_out, int out_size, void* d_ws, size_t ws_size,
                              hipStream_t stream) {
  (void)in_sizes; (void)n_in; (void)out_size; (void)ws_size;
  const float* x    = (const float*)d_in[0];
  const float* mask = (const float*)d_in[1];
  const float* Wq   = (const float*)d_in[2];
  const float* bq   = (const float*)d_in[3];
  const float* Wk   = (const float*)d_in[4];
  const float* Wv   = (const float*)d_in[5];
  const float* bv   = (const float*)d_in[6];
  const float* Wo   = (const float*)d_in[7];
  const float* bo   = (const float*)d_in[8];

  float* out = (float*)d_out;                                   // [B,T,D]
  float* qk  = (float*)d_out + (size_t)BATCH * T_SEQ * D_MODEL; // [B,H,T,T]

  // workspace layout (uint16 elements); total 28M elems = 56 MiB
  const size_t M1 = 1048576;
  uint16_t* w     = (uint16_t*)d_ws;
  uint16_t* xbf   = w;            // 4M  : x bf16
  uint16_t* wqb   = w + 4  * M1;  // 1M
  uint16_t* wkb   = w + 5  * M1;  // 1M
  uint16_t* wvb   = w + 6  * M1;  // 1M
  uint16_t* wob   = w + 7  * M1;  // 1M
  uint16_t* qbf   = w + 8  * M1;  // 4M  : q (scaled) bf16
  uint16_t* kbf   = w + 12 * M1;  // 4M  : k (scaled) bf16
  uint16_t* vbf   = w + 16 * M1;  // 4M  : v bf16
  uint16_t* vtb   = w + 20 * M1;  // 4M  : v transposed [B,H,DH,T]
  uint16_t* wvact = w + 24 * M1;  // 4M  : softmax@V result bf16

  const int nTok = BATCH * T_SEQ * D_MODEL;   // 4M
  const int nW   = D_MODEL * D_MODEL;         // 1M

  // 1) pack fp32 -> bf16
  pack_bf16<<<nTok / 256, 256, 0, stream>>>(x,  xbf, nTok);
  pack_bf16<<<nW   / 256, 256, 0, stream>>>(Wq, wqb, nW);
  pack_bf16<<<nW   / 256, 256, 0, stream>>>(Wk, wkb, nW);
  pack_bf16<<<nW   / 256, 256, 0, stream>>>(Wv, wvb, nW);
  pack_bf16<<<nW   / 256, 256, 0, stream>>>(Wo, wob, nW);

  // 2) QKV projections (NT GEMM, 64x64 tile per wave): 1024 waves -> 128 blocks
  const int M = BATCH * T_SEQ, N = D_MODEL, K = D_MODEL;
  const int gemmBlocks = (M / 64) * (N / 64) / 8;
  gemm_nt<false, true ><<<gemmBlocks, 256, 0, stream>>>(xbf, wqb, bq, ATTN_SCALE, qbf, M, N, K);
  gemm_nt<false, false><<<gemmBlocks, 256, 0, stream>>>(xbf, wkb, nullptr, ATTN_SCALE, kbf, M, N, K);
  gemm_nt<false, true ><<<gemmBlocks, 256, 0, stream>>>(xbf, wvb, bv, 1.0f, vbf, M, N, K);

  // 3) scores + mask -> qk (fp32, second output region)
  attn_scores<<<dim3(32, BATCH * NHEAD), 256, 0, stream>>>(qbf, kbf, mask, qk);

  // 4) transpose v
  transpose_v<<<nTok / 256, 256, 0, stream>>>(vbf, vtb);

  // 5) fused softmax + P@V : 4096 waves -> 512 blocks
  softmax_wv<<<512, 256, 0, stream>>>(qk, vtb, wvact);

  // 6) output projection -> out (fp32)
  gemm_nt<true, true><<<gemmBlocks, 256, 0, stream>>>(wvact, wob, bo, 1.0f, out, M, N, K);
}
